// RelPosMHA_11020886082228
// MI455X (gfx1250) — matrix operand order repaired and checked by reading, not compiled, not run on hardware
//
#include <hip/hip_runtime.h>
#include <hip/hip_bf16.h>

typedef __attribute__((ext_vector_type(16))) __bf16 v16bf;
typedef __attribute__((ext_vector_type(8)))  __bf16 v8bf;
typedef __attribute__((ext_vector_type(4)))  __bf16 v4bf;
typedef __attribute__((ext_vector_type(8)))  float  v8f;

#define NB   32
#define NT   512
#define ND   512
#define NH   8
#define NDK  64
#define NTOK (NB * NT)   // 16384

#define WMMA_BF16(A, Bm, C) \
  __builtin_amdgcn_wmma_f32_16x16x32_bf16(false, (A), false, (Bm), (short)0, (C), false, false)

__device__ __forceinline__ v16bf cat16(v8bf lo, v8bf hi) {
  v16bf r;
#pragma unroll
  for (int i = 0; i < 8; ++i) { r[i] = lo[i]; r[i + 8] = hi[i]; }
  return r;
}

// A-matrix 16x32 bf16 fragment. `row` points at this lane's matrix row (m = lane&15).
// Per ISA: lanes 0-15 hold K {0..7,16..23}; lanes 16-31 hold K {8..15,24..31}.
__device__ __forceinline__ v16bf afrag(const __bf16* row, int khig) {
  v8bf lo = *(const v8bf*)(row + khig * 8);
  v8bf hi = *(const v8bf*)(row + 16 + khig * 8);
  return cat16(lo, hi);
}

// B-matrix 32x16 bf16 fragment. `col` points at this lane's matrix column
// (n = lane&15), stored contiguously over K. Lanes 0-15: K {0..7,16..23}; 16-31: K {8..15,24..31}.
__device__ __forceinline__ v16bf bfrag(const __bf16* col, int khig) {
  const __bf16* p = col + khig * 8;
  return cat16(*(const v8bf*)p, *(const v8bf*)(p + 16));
}

// Async global->LDS copy of 16 bytes (per lane), tracked by ASYNCcnt.
__device__ __forceinline__ void async_cp_b128(const __bf16* gsrc, __bf16* ldst) {
  unsigned lofs = (unsigned)(size_t)ldst;               // LDS aperture: low 32 bits = offset
  unsigned long long g = (unsigned long long)(size_t)gsrc;
  asm volatile("global_load_async_to_lds_b128 %0, %1, off" :: "v"(lofs), "v"(g) : "memory");
}
__device__ __forceinline__ void async_wait0() {
  asm volatile("s_wait_asynccnt 0x0" ::: "memory");
}

// ---------------------------------------------------------------- converts
__global__ __launch_bounds__(256) void cvt4_f32_bf16(const float* __restrict__ s,
                                                     __bf16* __restrict__ d, int n4) {
  int i = blockIdx.x * blockDim.x + threadIdx.x;
  if (i < n4) {
    float4 f = ((const float4*)s)[i];
    v4bf o;
    o[0] = (__bf16)f.x; o[1] = (__bf16)f.y; o[2] = (__bf16)f.z; o[3] = (__bf16)f.w;
    *(v4bf*)(d + i * 4) = o;
  }
}

// ---------------------------------------------------------------- GEMM
// Y[m, j] = sum_k X[m,k] * W[j,k] + bias[j]    (nn.Linear: x @ W.T + b)
// Workgroup tile: M=128 (8 waves x 16), N=64, K-step 32 staged via ASYNC copies.
// mode 0: dst bf16 [B,H,T,DK] (Q/K)   mode 2: dst bf16 [B,H,DK,T] (V transposed)
// mode 3: dst f32  [B,T,D]            (output projection)
__global__ __launch_bounds__(256) void gemm_bf16_k(const __bf16* __restrict__ X,
                                                   const __bf16* __restrict__ W,
                                                   const float* __restrict__ bias,
                                                   void* __restrict__ dst, int mode) {
  __shared__ __bf16 xs[128 * 40];   // 128 rows x 32 halves, stride 40 (80B, 16B-aligned)
  __shared__ __bf16 wl[64 * 40];    // 64 rows x 32 halves

  const int tid  = threadIdx.x;
  const int lane = tid & 31;
  const int wv   = tid >> 5;        // 0..7
  const int m0   = blockIdx.x * 128;
  const int n0   = blockIdx.y * 64;
  const int khig = lane >> 4;
  const int col  = lane & 15;

  // per-thread staging assignments (constant across K loop)
  const int xrow = tid >> 1, xhh = (tid & 1) * 16;   // X: 128 rows x 32 halves
  const int wrow = tid >> 2, wq  = (tid & 3) * 8;    // W: 64 rows x 32 halves

  v8f acc[4] = {};

  for (int k0 = 0; k0 < ND; k0 += 32) {
    __syncthreads();                       // previous tile fully consumed
    {   // async stage X tile (2 x b128 per thread, TDM-free ASYNC path)
      const __bf16* src = X + (size_t)(m0 + xrow) * ND + k0 + xhh;
      __bf16* dstl = xs + xrow * 40 + xhh;
      async_cp_b128(src,     dstl);
      async_cp_b128(src + 8, dstl + 8);
      if (k0 + 32 < ND) __builtin_prefetch(src + 32, 0, 0);
    }
    {   // async stage W tile (1 x b128 per thread)
      async_cp_b128(W + (size_t)(n0 + wrow) * ND + k0 + wq, wl + wrow * 40 + wq);
    }
    async_wait0();                         // this wave's copies have landed in LDS
    __syncthreads();                       // all waves' copies visible

    v16bf a = afrag(xs + (wv * 16 + col) * 40, khig);
#pragma unroll
    for (int t = 0; t < 4; ++t) {
      v16bf bm = bfrag(wl + (t * 16 + col) * 40, khig);
      acc[t] = WMMA_BF16(a, bm, acc[t]);
    }
  }

#pragma unroll
  for (int t = 0; t < 4; ++t) {
    int j  = n0 + t * 16 + col;
    float bj = bias[j];
#pragma unroll
    for (int r = 0; r < 8; ++r) {
      int m = m0 + wv * 16 + r + khig * 8;
      float val = acc[t][r] + bj;
      if (mode == 3) {
        ((float*)dst)[(size_t)m * ND + j] = val;
      } else {
        int b = m >> 9, tt = m & 511;
        int h = j >> 6, dk = j & 63;
        if (mode == 2)
          ((__bf16*)dst)[(((size_t)(b * NH + h) * NDK + dk) * NT) + tt] = (__bf16)val;
        else
          ((__bf16*)dst)[(((size_t)(b * NH + h) * NT + tt) * NDK) + dk] = (__bf16)val;
      }
    }
  }
}

// ---------------------------------------------------------------- flash attention
// Q,K: bf16 [B,H,T,DK]; Vt: bf16 [B,H,DK,T]; O: bf16 [B,T,D]
// One wave handles 16 query rows; 8 waves per block; grid = B*H*(T/128).
__global__ __launch_bounds__(256) void attn_k(const __bf16* __restrict__ Q,
                                              const __bf16* __restrict__ K,
                                              const __bf16* __restrict__ Vt,
                                              const float* __restrict__ rel,
                                              const unsigned char* __restrict__ kpm,
                                              __bf16* __restrict__ O) {
  __shared__ __bf16 pb[8 * 16 * 40];     // per-wave P transpose bounce (C->A layout)

  const int lane = threadIdx.x & 31;
  const int wv   = threadIdx.x >> 5;
  const int khig = lane >> 4;
  const int col  = lane & 15;

  const int bh   = blockIdx.x >> 2;          // b*H + h
  const int qblk = blockIdx.x & 3;
  const int b    = bh >> 3;
  const int h    = bh & 7;
  const int q0   = qblk * 128 + wv * 16;

  const __bf16* Qp = Q  + (size_t)bh * NT * NDK;
  const __bf16* Kp = K  + (size_t)bh * NT * NDK;
  const __bf16* Vp = Vt + (size_t)bh * NDK * NT;
  __bf16* pw = pb + wv * 16 * 40;

  // Preload Q fragments (16 x 64 = two 16x32 A-frags)
  const __bf16* qrow = Qp + (size_t)(q0 + col) * NDK;
  v16bf qa0 = afrag(qrow, khig);
  v16bf qa1 = afrag(qrow + 32, khig);

  v8f oacc[4] = {};
  float mrun[8], lsum[8];
#pragma unroll
  for (int r = 0; r < 8; ++r) { mrun[r] = -1e30f; lsum[r] = 0.0f; }

  for (int kt0 = 0; kt0 < NT; kt0 += 32) {
    // ---- S = Q @ K^T (two 16-key tiles)
    v8f s0 = {}, s1 = {};
    {
      const __bf16* kc0 = Kp + (size_t)(kt0 + col) * NDK;        // key tile 0, column=key
      s0 = WMMA_BF16(qa0, bfrag(kc0, khig), s0);
      s0 = WMMA_BF16(qa1, bfrag(kc0 + 32, khig), s0);
      const __bf16* kc1 = kc0 + 16 * NDK;                        // key tile 1
      s1 = WMMA_BF16(qa0, bfrag(kc1, khig), s1);
      s1 = WMMA_BF16(qa1, bfrag(kc1 + 32, khig), s1);
    }

    // ---- bias + mask + online softmax
    const int kp0 = kt0 + col;
    const bool msk0 = kpm[b * NT + kp0] != 0;
    const bool msk1 = kpm[b * NT + kp0 + 16] != 0;
#pragma unroll
    for (int r = 0; r < 8; ++r) {
      int qpos = q0 + r + khig * 8;
      float b0 = rel[(kp0 - qpos + (NT - 1)) * NH + h];
      float b1 = rel[(kp0 + 16 - qpos + (NT - 1)) * NH + h];
      float t0 = msk0 ? -1e30f : s0[r] * 0.125f + b0;
      float t1 = msk1 ? -1e30f : s1[r] * 0.125f + b1;
      float v = fmaxf(t0, t1);
      v = fmaxf(v, __shfl_xor(v, 1));
      v = fmaxf(v, __shfl_xor(v, 2));
      v = fmaxf(v, __shfl_xor(v, 4));
      v = fmaxf(v, __shfl_xor(v, 8));        // 16-lane group == row owners
      float mnew  = fmaxf(mrun[r], v);
      float alpha = __expf(mrun[r] - mnew);
      mrun[r] = mnew;
      float p0 = __expf(t0 - mnew);
      float p1 = __expf(t1 - mnew);
      float rs = p0 + p1;
      rs += __shfl_xor(rs, 1);
      rs += __shfl_xor(rs, 2);
      rs += __shfl_xor(rs, 4);
      rs += __shfl_xor(rs, 8);
      lsum[r] = lsum[r] * alpha + rs;
#pragma unroll
      for (int t = 0; t < 4; ++t) oacc[t][r] *= alpha;
      s0[r] = p0; s1[r] = p1;
    }

    // ---- P: C-layout -> A-layout via per-wave LDS bounce
#pragma unroll
    for (int r = 0; r < 8; ++r) {
      int row = r + khig * 8;
      pw[row * 40 + col]      = (__bf16)s0[r];
      pw[row * 40 + 16 + col] = (__bf16)s1[r];
    }
    asm volatile("s_wait_dscnt 0x0" ::: "memory");   // same-wave LDS RAW fence
    v16bf pa = afrag(pw + col * 40, khig);

    // ---- O += P @ V  (V stored transposed: column dk contiguous over keys)
#pragma unroll
    for (int t = 0; t < 4; ++t) {
      const __bf16* vc = Vp + (size_t)(t * 16 + col) * NT + kt0;
      oacc[t] = WMMA_BF16(pa, bfrag(vc, khig), oacc[t]);
    }
  }

  // ---- normalize + write O[b, t, h*64 + dk]
#pragma unroll
  for (int r = 0; r < 8; ++r) {
    float inv = 1.0f / lsum[r];
    int t = q0 + r + khig * 8;
#pragma unroll
    for (int a = 0; a < 4; ++a) {
      O[((size_t)(b * NT + t) * ND) + h * NDK + a * 16 + col] = (__bf16)(oacc[a][r] * inv);
    }
  }
}

// ---------------------------------------------------------------- launch
extern "C" void kernel_launch(void* const* d_in, const int* in_sizes, int n_in,
                              void* d_out, int out_size, void* d_ws, size_t ws_size,
                              hipStream_t stream) {
  const float* x   = (const float*)d_in[0];
  const float* Wq  = (const float*)d_in[1];
  const float* bq  = (const float*)d_in[2];
  const float* Wk  = (const float*)d_in[3];
  const float* bk  = (const float*)d_in[4];
  const float* Wv  = (const float*)d_in[5];
  const float* bv  = (const float*)d_in[6];
  const float* Wo  = (const float*)d_in[7];
  const float* bo  = (const float*)d_in[8];
  const float* rel = (const float*)d_in[9];
  const unsigned char* kpm = (const unsigned char*)d_in[10];
  float* out = (float*)d_out;

  // workspace layout (bf16 buffers)
  char* ws = (char*)d_ws;
  const size_t SZ_X  = (size_t)NTOK * ND * 2;      // 16 MB
  const size_t SZ_W  = (size_t)ND * ND * 2;        // 512 KB
  __bf16* xb  = (__bf16*)ws;
  __bf16* wqb = (__bf16*)(ws + SZ_X);
  __bf16* wkb = wqb + (size_t)ND * ND;
  __bf16* wvb = wkb + (size_t)ND * ND;
  __bf16* wob = wvb + (size_t)ND * ND;
  __bf16* qb  = (__bf16*)(ws + SZ_X + 4 * SZ_W);
  __bf16* kb  = qb + (size_t)NB * NH * NT * NDK;
  __bf16* vtb = kb + (size_t)NB * NH * NT * NDK;
  __bf16* ob  = vtb + (size_t)NB * NH * NT * NDK;
  (void)in_sizes; (void)n_in; (void)out_size; (void)ws_size;

  // 1) bf16 conversions
  cvt4_f32_bf16<<<(NTOK * ND / 4 + 255) / 256, 256, 0, stream>>>(x, xb, NTOK * ND / 4);
  cvt4_f32_bf16<<<(ND * ND / 4 + 255) / 256, 256, 0, stream>>>(Wq, wqb, ND * ND / 4);
  cvt4_f32_bf16<<<(ND * ND / 4 + 255) / 256, 256, 0, stream>>>(Wk, wkb, ND * ND / 4);
  cvt4_f32_bf16<<<(ND * ND / 4 + 255) / 256, 256, 0, stream>>>(Wv, wvb, ND * ND / 4);
  cvt4_f32_bf16<<<(ND * ND / 4 + 255) / 256, 256, 0, stream>>>(Wo, wob, ND * ND / 4);

  // 2) projections
  dim3 ggrid(NTOK / 128, ND / 64);
  gemm_bf16_k<<<ggrid, 256, 0, stream>>>(xb, wqb, bq, (void*)qb, 0);
  gemm_bf16_k<<<ggrid, 256, 0, stream>>>(xb, wkb, bk, (void*)kb, 0);
  gemm_bf16_k<<<ggrid, 256, 0, stream>>>(xb, wvb, bv, (void*)vtb, 2);

  // 3) flash attention with relative-position bias + key padding mask
  attn_k<<<NB * NH * (NT / 128), 256, 0, stream>>>(qb, kb, vtb, rel, kpm, ob);

  // 4) output projection (f32 result)
  gemm_bf16_k<<<ggrid, 256, 0, stream>>>(ob, wob, bo, (void*)out, 3);
}
